// DeformableTransformer_8031588844213
// MI455X (gfx1250) — matrix operand order repaired and checked by reading, not compiled, not run on hardware
//
#include <hip/hip_runtime.h>

// ===========================================================================
// Deformable-DETR forward for MI455X (gfx1250, wave32, WMMA).
// All dense linears run on v_wmma_f32_16x16x32_bf16 (bf16 in, f32 acc).
// Each wave owns a 32x64 C tile: 8 WMMAs per K-step, A frags reused 4x.
// Out-of-range rows/cols load from clamped addresses; their garbage only
// reaches accumulator elements that the (guarded) epilogue never stores.
// ===========================================================================

typedef __attribute__((ext_vector_type(16))) __bf16 v16bf;
typedef __attribute__((ext_vector_type(8)))  float  v8f;

#define GF_RELU  1
#define GF_RESID 2

#define LV_TOT  21760      // 128*128 + 64*64 + 32*32 + 16*16
#define M_ENC   43520      // B(=2) * LV_TOT
#define NQ_DEC  300
#define M_DEC   600        // B * NQ_DEC

// ---------------------------------------------------------------------------
// Fragment builders (CDNA5 ISA 7.12.2 VGPR layouts):
//   A 16x32 bf16 : lane l (row M=l&15): e<8 -> K=k0+8*(l>=16)+e,
//                                       e>=8 -> K=k0+16+8*(l>=16)+(e-8)
//   B 32x16 bf16 : lane l (col N=l&15): same K mapping as A
//   C 16x16 f32  : VGPR r: lanes 0-15 -> M=r, lanes 16-31 -> M=r+8, N=l&15
// Native (__bf16) fptrunc -> v_cvt_pk_bf16_f32 (RTNE).
// ---------------------------------------------------------------------------
__device__ __forceinline__ v16bf a_frag(const float* __restrict__ row,
                                        int k0, int half) {
  const float* p = row + k0 + half * 8;
  const float4 a0 = *(const float4*)(p);
  const float4 a1 = *(const float4*)(p + 4);
  const float4 a2 = *(const float4*)(p + 16);
  const float4 a3 = *(const float4*)(p + 20);
  const float f[16] = {a0.x,a0.y,a0.z,a0.w, a1.x,a1.y,a1.z,a1.w,
                       a2.x,a2.y,a2.z,a2.w, a3.x,a3.y,a3.z,a3.w};
  v16bf v;
#pragma unroll
  for (int e = 0; e < 16; ++e) v[e] = (__bf16)f[e];
  return v;
}

__device__ __forceinline__ v16bf b_frag(const float* __restrict__ row,
                                        int k0, int half) {
  const float* p = row + k0 + half * 8;
  const float4 b0 = *(const float4*)(p);
  const float4 b1 = *(const float4*)(p + 4);
  const float4 b2 = *(const float4*)(p + 16);
  const float4 b3 = *(const float4*)(p + 20);
  const float f[16] = {b0.x,b0.y,b0.z,b0.w, b1.x,b1.y,b1.z,b1.w,
                       b2.x,b2.y,b2.z,b2.w, b3.x,b3.y,b3.z,b3.w};
  v16bf v;
#pragma unroll
  for (int e = 0; e < 16; ++e) v[e] = (__bf16)f[e];
  return v;
}

// ---------------------------------------------------------------------------
// WMMA GEMM: C[M,N] = A[M,K] @ W[N,K]^T + bias (+residual R) (optional ReLU)
// One wave per 32x64 C tile (2 M-subtiles x 4 N-subtiles).
// ---------------------------------------------------------------------------
__global__ __launch_bounds__(32) void gemm32x64_wmma(
    const float* __restrict__ A, const float* __restrict__ W,
    const float* __restrict__ bias, const float* __restrict__ R,
    float* __restrict__ C, int M, int N, int K, int flags)
{
  const int lane = threadIdx.x & 31;
  const int nt = blockIdx.x, mt = blockIdx.y;
  const int half = lane >> 4;
  const int l15  = lane & 15;

  const int m0 = mt * 32 + l15;
  const int m1 = m0 + 16;
  const float* arow0 = A + (long)(m0 < M ? m0 : M - 1) * K;
  const float* arow1 = A + (long)(m1 < M ? m1 : M - 1) * K;

  int  nn[4];  bool nv[4];  const float* wrow[4];
#pragma unroll
  for (int j = 0; j < 4; ++j) {
    nn[j] = nt * 64 + j * 16 + l15;
    nv[j] = (nn[j] < N);
    wrow[j] = W + (long)(nv[j] ? nn[j] : N - 1) * K;
  }

  v8f acc[2][4];
#pragma unroll
  for (int i = 0; i < 2; ++i)
#pragma unroll
    for (int j = 0; j < 4; ++j) acc[i][j] = (v8f){};

  for (int k0 = 0; k0 < K; k0 += 32) {
    if (k0 + 32 < K) {                       // global_prefetch_b8 next K-slab
      __builtin_prefetch(arow0 + k0 + 32, 0, 3);
      __builtin_prefetch(arow1 + k0 + 32, 0, 3);
      __builtin_prefetch(wrow[0] + k0 + 32, 0, 3);
    }
    const v16bf va0 = a_frag(arow0, k0, half);
    const v16bf va1 = a_frag(arow1, k0, half);
#pragma unroll
    for (int j = 0; j < 4; ++j) {
      if (nt * 64 + j * 16 < N) {            // wave-uniform: EXEC stays all-1s
        const v16bf vb = b_frag(wrow[j], k0, half);
        acc[0][j] = __builtin_amdgcn_wmma_f32_16x16x32_bf16(
            false, va0, false, vb, (short)0, acc[0][j], false, false);
        acc[1][j] = __builtin_amdgcn_wmma_f32_16x16x32_bf16(
            false, va1, false, vb, (short)0, acc[1][j], false, false);
      }
    }
  }

#pragma unroll
  for (int j = 0; j < 4; ++j) {
    if (nt * 64 + j * 16 >= N) continue;     // uniform
    const float bvl = (nv[j] && bias) ? bias[nn[j]] : 0.0f;
#pragma unroll
    for (int i = 0; i < 2; ++i) {
#pragma unroll
      for (int r = 0; r < 8; ++r) {
        const int mr = mt * 32 + i * 16 + r + half * 8;
        if (nv[j] && mr < M) {
          float v = acc[i][j][r] + bvl;
          if (flags & GF_RESID) v += R[(long)mr * N + nn[j]];
          if (flags & GF_RELU)  v = fmaxf(v, 0.0f);
          C[(long)mr * N + nn[j]] = v;
        }
      }
    }
  }
}

// ---------------------------------------------------------------------------
// Elementwise helpers
// ---------------------------------------------------------------------------
__global__ void add_k(const float* __restrict__ a, const float* __restrict__ b,
                      float* __restrict__ o, long n) {
  long i = (long)blockIdx.x * blockDim.x + threadIdx.x;
  if (i < n) o[i] = a[i] + b[i];
}

__global__ void copy_f32(const float* __restrict__ a, float* __restrict__ o, long n) {
  long i = (long)blockIdx.x * blockDim.x + threadIdx.x;
  if (i < n) o[i] = a[i];
}

__global__ void sigmoid_k(float* __restrict__ p, long n) {
  long i = (long)blockIdx.x * blockDim.x + threadIdx.x;
  if (i < n) p[i] = 1.0f / (1.0f + __expf(-p[i]));
}

// [B,256,H,W] -> [B,Lv,256] (+ pos + level_embed into P)
__global__ void flatten_level(const float* __restrict__ src,
                              const float* __restrict__ pos,
                              const float* __restrict__ lemb,
                              float* __restrict__ X, float* __restrict__ P,
                              int HW, int start) {
  long i = (long)blockIdx.x * blockDim.x + threadIdx.x;
  long total = (long)2 * HW * 256;
  if (i >= total) return;
  int c = (int)(i % 256);
  long rest = i / 256;
  int sp = (int)(rest % HW);
  int b  = (int)(rest / HW);
  long si = ((long)b * 256 + c) * HW + sp;
  long di = ((long)b * LV_TOT + start + sp) * 256 + c;
  X[di] = src[si];
  P[di] = pos[si] + lemb[c];
}

// query_embed [300,512] -> qp [2,300,256], tgt [2,300,256]
__global__ void split_query(const float* __restrict__ qe,
                            float* __restrict__ qp, float* __restrict__ tgt) {
  int i = blockIdx.x * blockDim.x + threadIdx.x;
  if (i >= 2 * 300 * 256) return;
  int c  = i % 256;
  int qi = (i / 256) % 300;
  qp[i]  = qe[qi * 512 + c];
  tgt[i] = qe[qi * 512 + 256 + c];
}

// Encoder reference points (valid_ratios==1 since masks are all-false):
// ref[i] = ((px+0.5)/W_l, (py+0.5)/H_l), identical for every level.
__global__ void enc_refpoints(float* __restrict__ ref) {
  int i = blockIdx.x * blockDim.x + threadIdx.x;
  if (i >= LV_TOT) return;
  int S, W, H;
  if      (i < 16384) { S = 0;     W = 128; H = 128; }
  else if (i < 20480) { S = 16384; W = 64;  H = 64;  }
  else if (i < 21504) { S = 20480; W = 32;  H = 32;  }
  else                { S = 21504; W = 16;  H = 16;  }
  int sp = i - S;
  ref[i * 2 + 0] = ((sp % W) + 0.5f) / (float)W;
  ref[i * 2 + 1] = ((sp / W) + 0.5f) / (float)H;
}

// softmax over groups of 16 (attention weights), in place
__global__ void softmax16(float* __restrict__ p, long groups) {
  long g = (long)blockIdx.x * blockDim.x + threadIdx.x;
  if (g >= groups) return;
  float* q = p + g * 16;
  float v[16], mx = -3.0e38f;
#pragma unroll
  for (int i = 0; i < 16; ++i) { v[i] = q[i]; mx = fmaxf(mx, v[i]); }
  float s = 0.f;
#pragma unroll
  for (int i = 0; i < 16; ++i) { v[i] = __expf(v[i] - mx); s += v[i]; }
  const float inv = 1.0f / s;
#pragma unroll
  for (int i = 0; i < 16; ++i) q[i] = v[i] * inv;
}

// LayerNorm over last dim 256; one 256-thread block per row; in/out may alias
__global__ __launch_bounds__(256) void layernorm256(
    const float* __restrict__ in, const float* __restrict__ g,
    const float* __restrict__ b, float* __restrict__ out) {
  __shared__ float red[16];
  const long row = blockIdx.x;
  const int t = threadIdx.x;
  float x = in[row * 256 + t];
  float s = x, s2 = x * x;
#pragma unroll
  for (int o = 16; o; o >>= 1) { s += __shfl_xor(s, o, 32); s2 += __shfl_xor(s2, o, 32); }
  if ((t & 31) == 0) { red[t >> 5] = s; red[8 + (t >> 5)] = s2; }
  __syncthreads();
  float ts = 0.f, ts2 = 0.f;
#pragma unroll
  for (int i = 0; i < 8; ++i) { ts += red[i]; ts2 += red[8 + i]; }
  const float mu  = ts * (1.0f / 256.0f);
  const float var = ts2 * (1.0f / 256.0f) - mu * mu;
  out[row * 256 + t] = (x - mu) * rsqrtf(var + 1e-5f) * g[t] + b[t];
}

// ---------------------------------------------------------------------------
// MS-deformable attention sampling core.
// value [2,Lv,8,32], off [2,Nq,8,4,4,2], aw(softmaxed) [2,Nq,8,4,4],
// ref [.,2] (refBS=0: shared across batch), out [2,Nq,256] (head-major).
// One thread per (b, q, head); float4 channel loads.
// ---------------------------------------------------------------------------
__global__ void msda_sample(const float* __restrict__ value,
                            const float* __restrict__ off,
                            const float* __restrict__ aw,
                            const float* __restrict__ ref,
                            float* __restrict__ out,
                            int Nq, int refBS) {
  const int LW[4] = {128, 64, 32, 16};
  const int LH[4] = {128, 64, 32, 16};
  const int LS[4] = {0, 16384, 20480, 21504};

  long tid = (long)blockIdx.x * blockDim.x + threadIdx.x;
  long total = (long)2 * Nq * 8;
  if (tid >= total) return;
  const int h = (int)(tid % 8);
  const long bq = tid / 8;
  const int q = (int)(bq % Nq);
  const int b = (int)(bq / Nq);

  const float* rp = ref + (long)(b * refBS + q) * 2;
  const float rx = rp[0], ry = rp[1];
  const float* op = off + (bq * 8 + h) * 32;   // 4*4*2 per head
  const float* ap = aw  + (bq * 8 + h) * 16;   // 4*4 per head

  float acc[32];
#pragma unroll
  for (int d = 0; d < 32; ++d) acc[d] = 0.f;

#pragma unroll
  for (int l = 0; l < 4; ++l) {
    const int Wl = LW[l], Hl = LH[l];
    const float* vbase = value + (((long)b * LV_TOT + LS[l]) * 8 + h) * 32;
    const float fw = (float)Wl, fh = (float)Hl;
    for (int pt = 0; pt < 4; ++pt) {
      const float a = ap[l * 4 + pt];
      // (rx + ox/W)*W - 0.5 == rx*W + ox - 0.5  (no divide)
      const float x = rx * fw + op[(l * 4 + pt) * 2 + 0] - 0.5f;
      const float y = ry * fh + op[(l * 4 + pt) * 2 + 1] - 0.5f;
      const float fx = floorf(x), fy = floorf(y);
      const int x0 = (int)fx, y0 = (int)fy;
      const float lx = x - fx, ly = y - fy;
      const float cw[4] = {(1.f - ly) * (1.f - lx) * a, (1.f - ly) * lx * a,
                           ly * (1.f - lx) * a,         ly * lx * a};
      const int cx[4] = {x0, x0 + 1, x0,     x0 + 1};
      const int cy[4] = {y0, y0,     y0 + 1, y0 + 1};
#pragma unroll
      for (int c = 0; c < 4; ++c) {
        const int xi = cx[c], yi = cy[c];
        if (xi >= 0 && xi < Wl && yi >= 0 && yi < Hl) {
          const float w = cw[c];
          const float* vp = vbase + ((long)yi * Wl + xi) * 256; // 8 heads * 32
#pragma unroll
          for (int d = 0; d < 32; d += 4) {
            const float4 v4 = *(const float4*)(vp + d);
            acc[d + 0] += w * v4.x; acc[d + 1] += w * v4.y;
            acc[d + 2] += w * v4.z; acc[d + 3] += w * v4.w;
          }
        }
      }
    }
  }
  float* o = out + bq * 256 + h * 32;
#pragma unroll
  for (int d = 0; d < 32; d += 4) {
    *(float4*)(o + d) = make_float4(acc[d], acc[d + 1], acc[d + 2], acc[d + 3]);
  }
}

// ---------------------------------------------------------------------------
// Decoder self-attention: one wave per (query, head, batch).
// QK [2,300,512] (cols 0..255 = Q head-major, 256..511 = K), V [2,300,256].
// ---------------------------------------------------------------------------
__global__ __launch_bounds__(32) void dec_attn(const float* __restrict__ QK,
                                               const float* __restrict__ V,
                                               float* __restrict__ O) {
  const int qi = blockIdx.x, h = blockIdx.y, b = blockIdx.z;
  const int lane = threadIdx.x & 31;
  const int N = 300;
  const float scale = 0.1767766952966369f; // 1/sqrt(32)

  const float* Qp = QK + ((long)(b * N + qi)) * 512 + h * 32;
  float q[32];
#pragma unroll
  for (int d = 0; d < 32; d += 4) {
    const float4 v4 = *(const float4*)(Qp + d);
    q[d] = v4.x * scale; q[d + 1] = v4.y * scale;
    q[d + 2] = v4.z * scale; q[d + 3] = v4.w * scale;
  }

  float sc[10];
  float mx = -3.0e38f;
#pragma unroll
  for (int t = 0; t < 10; ++t) {
    const int k = lane + t * 32;
    float s = -3.0e38f;
    if (k < N) {
      const float* Kp = QK + ((long)(b * N + k)) * 512 + 256 + h * 32;
      float dot = 0.f;
#pragma unroll
      for (int d = 0; d < 32; d += 4) {
        const float4 v4 = *(const float4*)(Kp + d);
        dot += q[d] * v4.x + q[d + 1] * v4.y + q[d + 2] * v4.z + q[d + 3] * v4.w;
      }
      s = dot;
    }
    sc[t] = s;
    mx = fmaxf(mx, s);
  }
#pragma unroll
  for (int o = 16; o; o >>= 1) mx = fmaxf(mx, __shfl_xor(mx, o, 32));
  float sum = 0.f;
#pragma unroll
  for (int t = 0; t < 10; ++t) { sc[t] = __expf(sc[t] - mx); sum += sc[t]; }
#pragma unroll
  for (int o = 16; o; o >>= 1) sum += __shfl_xor(sum, o, 32);
  const float inv = 1.0f / sum;

  float acc[32];
#pragma unroll
  for (int d = 0; d < 32; ++d) acc[d] = 0.f;
#pragma unroll
  for (int t = 0; t < 10; ++t) {
    const int k = lane + t * 32;
    if (k < N) {
      const float w = sc[t] * inv;
      const float* Vp = V + ((long)(b * N + k)) * 256 + h * 32;
#pragma unroll
      for (int d = 0; d < 32; d += 4) {
        const float4 v4 = *(const float4*)(Vp + d);
        acc[d] += w * v4.x; acc[d + 1] += w * v4.y;
        acc[d + 2] += w * v4.z; acc[d + 3] += w * v4.w;
      }
    }
  }
  __shared__ float lds[32 * 32];
#pragma unroll
  for (int d = 0; d < 32; ++d) lds[lane * 32 + d] = acc[d];
  __syncthreads();
  float o = 0.f;
#pragma unroll
  for (int r = 0; r < 32; ++r) o += lds[r * 32 + lane];
  O[((long)(b * N + qi)) * 256 + h * 32 + lane] = o;
}

// ===========================================================================
// Host orchestration
// ===========================================================================
extern "C" void kernel_launch(void* const* d_in, const int* in_sizes, int n_in,
                              void* d_out, int out_size, void* d_ws, size_t ws_size,
                              hipStream_t stream) {
  (void)in_sizes; (void)n_in; (void)out_size; (void)ws_size;
  auto F = [&](int i) { return (const float*)d_in[i]; };

  // ---- input mapping: JAX pytree flatten (dict keys sorted alphabetically) --
  // 0..3: mask0..mask3 (all-false by construction -> valid_ratios == 1)
  struct Msda { const float *aw_b,*aw_w,*off_b,*off_w,*op_b,*op_w,*vp_b,*vp_w; };
  struct DecL { Msda ca; const float *l1b,*l1w,*l2b,*l2w,*n1b,*n1g,*n2b,*n2g,
                *n3b,*n3g,*sib,*siw,*sob,*sow; } dec[6];
  struct EncL { const float *l1b,*l1w,*l2b,*l2w,*n1b,*n1g,*n2b,*n2g; Msda sa; } enc[6];
  int p = 4;
  for (int i = 0; i < 6; ++i) {   // params.dec[i]: ca, lin1, lin2, norm1..3, sa_in, sa_out
    DecL& d = dec[i];
    d.ca.aw_b = F(p++); d.ca.aw_w = F(p++);
    d.ca.off_b = F(p++); d.ca.off_w = F(p++);
    d.ca.op_b = F(p++); d.ca.op_w = F(p++);
    d.ca.vp_b = F(p++); d.ca.vp_w = F(p++);
    d.l1b = F(p++); d.l1w = F(p++); d.l2b = F(p++); d.l2w = F(p++);
    d.n1b = F(p++); d.n1g = F(p++); d.n2b = F(p++); d.n2g = F(p++);
    d.n3b = F(p++); d.n3g = F(p++);
    d.sib = F(p++); d.siw = F(p++); d.sob = F(p++); d.sow = F(p++);
  }
  for (int i = 0; i < 6; ++i) {   // params.enc[i]: lin1, lin2, norm1, norm2, sa
    EncL& e = enc[i];
    e.l1b = F(p++); e.l1w = F(p++); e.l2b = F(p++); e.l2w = F(p++);
    e.n1b = F(p++); e.n1g = F(p++); e.n2b = F(p++); e.n2g = F(p++);
    e.sa.aw_b = F(p++); e.sa.aw_w = F(p++);
    e.sa.off_b = F(p++); e.sa.off_w = F(p++);
    e.sa.op_b = F(p++); e.sa.op_w = F(p++);
    e.sa.vp_b = F(p++); e.sa.vp_w = F(p++);
  }
  const float* level_embed = F(p++);
  const float* ref_b = F(p++);
  const float* ref_w = F(p++);
  const float* posL[4]; for (int i = 0; i < 4; ++i) posL[i] = F(p++);
  const float* query_embed = F(p++);
  const float* srcL[4]; for (int i = 0; i < 4; ++i) srcL[i] = F(p++);

  // ---- workspace layout (floats, 256B-aligned chunks; ~330 MB total) ------
  float* wsf = (float*)d_ws;
  long off = 0;
  auto alloc = [&](long nf) { float* q = wsf + off; off += (nf + 63) & ~63L; return q; };
  float* X    = alloc((long)M_ENC * 256);   // encoder/decoder stream ("memory")
  float* POS  = alloc((long)M_ENC * 256);
  float* QB   = alloc((long)M_ENC * 256);   // x + pos
  float* VP   = alloc((long)M_ENC * 256);   // value proj / big temp
  float* SAMP = alloc((long)M_ENC * 256);
  float* OFFB = alloc((long)M_ENC * 256);
  float* AWB  = alloc((long)M_ENC * 128);
  float* FFNH = alloc((long)4352 * 1024);   // FFN hidden, 10 row-chunks
  float* REF0 = alloc((long)LV_TOT * 2);
  float* QP   = alloc((long)M_DEC * 256);
  float* TGT  = alloc((long)M_DEC * 256);
  float* Q2   = alloc((long)M_DEC * 256);
  float* DT   = alloc((long)M_DEC * 256);
  float* QKB  = alloc((long)M_DEC * 512);
  float* VB   = alloc((long)M_DEC * 256);
  float* OB   = alloc((long)M_DEC * 256);
  float* DOFF = alloc((long)M_DEC * 256);
  float* DAW  = alloc((long)M_DEC * 128);
  float* DSMP = alloc((long)M_DEC * 256);
  float* DFFN = alloc((long)M_DEC * 1024);
  float* REFQ = alloc((long)M_DEC * 2);

  auto cdiv = [](long a, long b) { return (unsigned)((a + b - 1) / b); };
  auto gemm = [&](const float* A, const float* W, const float* bias,
                  const float* R, float* C, int M, int N, int K, int flags) {
    dim3 g((N + 63) / 64, (M + 31) / 32);
    gemm32x64_wmma<<<g, dim3(32), 0, stream>>>(A, W, bias, R, C, M, N, K, flags);
  };

  // ---- flatten + pos + level embedding -----------------------------------
  const int HWs[4]    = {16384, 4096, 1024, 256};
  const int starts[4] = {0, 16384, 20480, 21504};
  for (int l = 0; l < 4; ++l) {
    long tot = (long)2 * HWs[l] * 256;
    flatten_level<<<cdiv(tot, 256), 256, 0, stream>>>(
        srcL[l], posL[l], level_embed + l * 256, X, POS, HWs[l], starts[l]);
  }
  enc_refpoints<<<cdiv(LV_TOT, 256), 256, 0, stream>>>(REF0);

  // ---- encoder -----------------------------------------------------------
  const long NE = (long)M_ENC * 256;
  for (int L = 0; L < 6; ++L) {
    EncL& e = enc[L];
    add_k<<<cdiv(NE, 256), 256, 0, stream>>>(X, POS, QB, NE);              // q = x+pos
    gemm(X,  e.sa.vp_w,  e.sa.vp_b,  nullptr, VP,   M_ENC, 256, 256, 0);  // value
    gemm(QB, e.sa.off_w, e.sa.off_b, nullptr, OFFB, M_ENC, 256, 256, 0);  // offsets
    gemm(QB, e.sa.aw_w,  e.sa.aw_b,  nullptr, AWB,  M_ENC, 128, 256, 0);  // weights
    softmax16<<<cdiv((long)M_ENC * 8, 256), 256, 0, stream>>>(AWB, (long)M_ENC * 8);
    msda_sample<<<cdiv((long)M_ENC * 8, 256), 256, 0, stream>>>(
        VP, OFFB, AWB, REF0, SAMP, LV_TOT, /*refBS=*/0);
    gemm(SAMP, e.sa.op_w, e.sa.op_b, X, VP, M_ENC, 256, 256, GF_RESID);    // x + msda
    layernorm256<<<M_ENC, 256, 0, stream>>>(VP, e.n1g, e.n1b, X);
    for (int c = 0; c < 10; ++c) {                                         // FFN chunks
      long ro = (long)c * 4352;
      gemm(X + ro * 256, e.l1w, e.l1b, nullptr, FFNH, 4352, 1024, 256, GF_RELU);
      gemm(FFNH, e.l2w, e.l2b, X + ro * 256, VP + ro * 256, 4352, 256, 1024, GF_RESID);
    }
    layernorm256<<<M_ENC, 256, 0, stream>>>(VP, e.n2g, e.n2b, X);
  }
  // X now holds `memory`.

  // ---- decoder -----------------------------------------------------------
  split_query<<<cdiv((long)M_DEC * 256, 256), 256, 0, stream>>>(query_embed, QP, TGT);
  gemm(QP, ref_w, ref_b, nullptr, REFQ, M_DEC, 2, 256, 0);
  sigmoid_k<<<cdiv((long)M_DEC * 2, 256), 256, 0, stream>>>(REFQ, (long)M_DEC * 2);

  const long ND = (long)M_DEC * 256;
  for (int L = 0; L < 6; ++L) {
    DecL& d = dec[L];
    // self attention
    add_k<<<cdiv(ND, 256), 256, 0, stream>>>(TGT, QP, Q2, ND);
    gemm(Q2,  d.siw,             d.sib,       nullptr, QKB, M_DEC, 512, 256, 0);
    gemm(TGT, d.siw + 512 * 256, d.sib + 512, nullptr, VB,  M_DEC, 256, 256, 0);
    dec_attn<<<dim3(NQ_DEC, 8, 2), dim3(32), 0, stream>>>(QKB, VB, OB);
    gemm(OB, d.sow, d.sob, TGT, DT, M_DEC, 256, 256, GF_RESID);
    layernorm256<<<M_DEC, 256, 0, stream>>>(DT, d.n2g, d.n2b, TGT);
    // cross deformable attention over memory
    add_k<<<cdiv(ND, 256), 256, 0, stream>>>(TGT, QP, Q2, ND);
    gemm(X,  d.ca.vp_w,  d.ca.vp_b,  nullptr, VP,   M_ENC, 256, 256, 0);
    gemm(Q2, d.ca.off_w, d.ca.off_b, nullptr, DOFF, M_DEC, 256, 256, 0);
    gemm(Q2, d.ca.aw_w,  d.ca.aw_b,  nullptr, DAW,  M_DEC, 128, 256, 0);
    softmax16<<<cdiv((long)M_DEC * 8, 256), 256, 0, stream>>>(DAW, (long)M_DEC * 8);
    msda_sample<<<cdiv((long)M_DEC * 8, 256), 256, 0, stream>>>(
        VP, DOFF, DAW, REFQ, DSMP, NQ_DEC, /*refBS=*/NQ_DEC);
    gemm(DSMP, d.ca.op_w, d.ca.op_b, TGT, DT, M_DEC, 256, 256, GF_RESID);
    layernorm256<<<M_DEC, 256, 0, stream>>>(DT, d.n1g, d.n1b, TGT);
    // FFN
    gemm(TGT, d.l1w, d.l1b, nullptr, DFFN, M_DEC, 1024, 256, GF_RELU);
    gemm(DFFN, d.l2w, d.l2b, TGT, DT, M_DEC, 256, 1024, GF_RESID);
    layernorm256<<<M_DEC, 256, 0, stream>>>(DT, d.n3g, d.n3b, TGT);
  }

  // ---- outputs: (out, init_ref, ref) concatenated flat -------------------
  float* out = (float*)d_out;
  copy_f32<<<cdiv(153600, 256), 256, 0, stream>>>(TGT, out, 153600);
  copy_f32<<<cdiv(1200, 256), 256, 0, stream>>>(REFQ, out + 153600, 1200);
  copy_f32<<<cdiv(1200, 256), 256, 0, stream>>>(REFQ, out + 154800, 1200);
}
